// SRWCore_21406117003482
// MI455X (gfx1250) — matrix operand order repaired and checked by reading, not compiled, not run on hardware
//
#include <hip/hip_runtime.h>
#include <hip/hip_bf16.h>

// ---------------------------------------------------------------------------
// Shapes (fixed by the reference): B=16, N=4096, H=512, top-M=512, 8 heads x 64
// ---------------------------------------------------------------------------
#define BB   16
#define NN   4096
#define HH   512
#define MM   512
#define NHD  8
#define DH   64
#define ROWS (BB * MM)          // 8192 selected rows
#define EPS  1e-5f

typedef __attribute__((ext_vector_type(16))) _Float16 v16h;
typedef __attribute__((ext_vector_type(8)))  _Float16 v8h;
typedef __attribute__((ext_vector_type(4)))  _Float16 v4h;
typedef __attribute__((ext_vector_type(8)))  float    v8f;
typedef __attribute__((ext_vector_type(4)))  int      v4i;

// Async global->LDS (CDNA5); guarded so compilation never breaks.
#if defined(__has_builtin)
#if __has_builtin(__builtin_amdgcn_global_load_async_to_lds_b128) && \
    __has_builtin(__builtin_amdgcn_s_wait_asynccnt)
#define USE_ASYNC_LDS 1
#endif
#endif
#ifndef USE_ASYNC_LDS
#define USE_ASYNC_LDS 0
#endif

// ---------------------------------------------------------------------------
// WMMA fragment loaders (CDNA5 16x16x32 f16 layouts, wave32)
// A (16x32, MxK): lane L -> row m = L&15; kb=(L>=16)?8:0; two 16B runs at
//   K = kb..kb+7 and 16+kb..16+kb+7.
// B (32x16, KxN): lane L -> col n = L&15; kb=(L>=16)?8:0; two 16B runs at
//   K = kb..kb+7 and 16+kb..16+kb+7.
// C/D (16x16 f32): lane L -> col n = L&15; VGPR v -> row v + ((L>=16)?8:0).
// ---------------------------------------------------------------------------
__device__ __forceinline__ v16h load_a_frag(const _Float16* A, int lda, int m0,
                                            int k0, int lane) {
  int m  = lane & 15;
  int kb = (lane & 16) ? 8 : 0;
  const _Float16* p = A + (size_t)(m0 + m) * lda + k0 + kb;
  v8h lo = *(const v8h*)p;
  v8h hi = *(const v8h*)(p + 16);
  v16h a;
#pragma unroll
  for (int i = 0; i < 8; ++i) { a[i] = lo[i]; a[8 + i] = hi[i]; }
  return a;
}

__device__ __forceinline__ v16h load_b_frag(const _Float16* W, int ldw, int n0,
                                            int k0, int lane) {
  int n  = lane & 15;
  int kb = (lane & 16) ? 8 : 0;
  const _Float16* p = W + (size_t)(n0 + n) * ldw + k0 + kb;
  v8h lo = *(const v8h*)p;
  v8h hi = *(const v8h*)(p + 16);
  v16h b;
#pragma unroll
  for (int i = 0; i < 8; ++i) { b[i] = lo[i]; b[8 + i] = hi[i]; }
  return b;
}

__device__ __forceinline__ v8f wmma_f16(v16h a, v16h b, v8f c) {
  return __builtin_amdgcn_wmma_f32_16x16x32_f16(false, a, false, b,
                                                (short)0, c, false, false);
}

// ---------------------------------------------------------------------------
// 0) zero helper (pooled accumulator must be zeroed each launch)
// ---------------------------------------------------------------------------
__global__ void zero_kernel(float* p, int n) {
  int i = blockIdx.x * blockDim.x + threadIdx.x;
  if (i < n) p[i] = 0.f;
}

// ---------------------------------------------------------------------------
// 1) fused salience + pooled partial sums: ONE streaming pass over tokens.
//    One wave32 per 512-float row; 8 rows (same batch) per block.
//    Column sums go through an LDS f32 accumulator (ds_add_f32), then one
//    512-wide global atomicAdd per block.
// ---------------------------------------------------------------------------
__global__ void salience_pooled_kernel(const float* __restrict__ tokens,
                                       const float* __restrict__ g,
                                       const float* __restrict__ bb,
                                       const float* __restrict__ wv,
                                       const float* __restrict__ sb,
                                       unsigned* __restrict__ keys,
                                       float* __restrict__ pooled) {
  __shared__ float acc[HH];
  int tid = threadIdx.x, lane = tid & 31, w = tid >> 5;       // block 256 = 8 waves
  int row = blockIdx.x * 8 + w;                               // 0..65535
  int b = row >> 12;                                          // 4096 rows / batch
  for (int e = tid; e < HH; e += 256) acc[e] = 0.f;
  __syncthreads();

  const float* x = tokens + (size_t)row * HH;
  __builtin_prefetch(x + 8 * HH, 0, 0);
  float xs[16];
  float s = 0.f;
#pragma unroll
  for (int j = 0; j < 16; ++j) { float v = x[lane + 32 * j]; xs[j] = v; s += v; }
#pragma unroll
  for (int m = 16; m >= 1; m >>= 1) s += __shfl_xor(s, m, 32);
  float mean = s * (1.f / (float)HH);
  float s2 = 0.f;
#pragma unroll
  for (int j = 0; j < 16; ++j) { float d = xs[j] - mean; s2 += d * d; }
#pragma unroll
  for (int m = 16; m >= 1; m >>= 1) s2 += __shfl_xor(s2, m, 32);
  float r = rsqrtf(s2 * (1.f / (float)HH) + EPS);
  float dot = 0.f;
#pragma unroll
  for (int j = 0; j < 16; ++j) {
    int hh = lane + 32 * j;
    dot += ((xs[j] - mean) * r * g[hh] + bb[hh]) * wv[hh];
    atomicAdd(&acc[hh], xs[j]);                               // LDS ds_add_f32
  }
#pragma unroll
  for (int m = 16; m >= 1; m >>= 1) dot += __shfl_xor(dot, m, 32);
  if (lane == 0) {
    unsigned u = __float_as_uint(dot + sb[0]);
    keys[row] = (u & 0x80000000u) ? ~u : (u | 0x80000000u);
  }
  __syncthreads();
  for (int e = tid; e < HH; e += 256) atomicAdd(&pooled[b * HH + e], acc[e]);
}

// ---------------------------------------------------------------------------
// 2) gate: sigmoid(LN(pooled/N) . gate_w + gate_b), one block per batch
// ---------------------------------------------------------------------------
__global__ void gate_kernel(const float* __restrict__ pooled,
                            const float* __restrict__ g, const float* __restrict__ bb,
                            const float* __restrict__ gw, const float* __restrict__ gb,
                            float* gate_prob) {
  __shared__ float red[HH];
  int b = blockIdx.x, tid = threadIdx.x;                      // block 512
  float x = pooled[b * HH + tid] * (1.f / (float)NN);
  red[tid] = x; __syncthreads();
  for (int s = 256; s > 0; s >>= 1) { if (tid < s) red[tid] += red[tid + s]; __syncthreads(); }
  float mean = red[0] * (1.f / (float)HH);
  __syncthreads();
  float d = x - mean;
  red[tid] = d * d; __syncthreads();
  for (int s = 256; s > 0; s >>= 1) { if (tid < s) red[tid] += red[tid + s]; __syncthreads(); }
  float var = red[0] * (1.f / (float)HH);
  __syncthreads();
  float r = rsqrtf(var + EPS);
  red[tid] = (d * r * g[tid] + bb[tid]) * gw[tid]; __syncthreads();
  for (int s = 256; s > 0; s >>= 1) { if (tid < s) red[tid] += red[tid + s]; __syncthreads(); }
  if (tid == 0) {
    float z = red[0] + gb[0];
    gate_prob[b] = 1.f / (1.f + __expf(-z));
  }
}

// ---------------------------------------------------------------------------
// 3) top-512 of 4096 per batch: binary search on monotone-uint threshold +
//    deterministic scan compaction. Also writes inverse map inv[b][n] = slot
//    (-1 if unselected); MHA+LN are permutation-equivariant so order is free.
// ---------------------------------------------------------------------------
__global__ __launch_bounds__(1024) void topk_kernel(const unsigned* __restrict__ keys,
                                                    int* __restrict__ sidx,
                                                    int* __restrict__ inv) {
  __shared__ unsigned sk[NN];
  __shared__ int scnt;
  __shared__ int sscan[1024];
  int b = blockIdx.x, tid = threadIdx.x;                      // block 1024
  for (int j = tid; j < NN; j += 1024) {
    sk[j] = keys[b * NN + j];
    inv[b * NN + j] = -1;        // same thread owns j in compaction loops below
  }
  __syncthreads();
  // largest T with count(key >= T) >= MM  (lo/hi uniform: derived from shared)
  unsigned lo = 0u, hi = 0xFFFFFFFFu;
  while (lo < hi) {
    unsigned mid = lo + ((hi - lo + 1u) >> 1);
    if (tid == 0) scnt = 0;
    __syncthreads();
    int c = 0;
    for (int j = tid; j < NN; j += 1024) c += (sk[j] >= mid) ? 1 : 0;
    atomicAdd(&scnt, c);
    __syncthreads();
    int total = scnt;
    __syncthreads();
    if (total >= MM) lo = mid; else hi = mid - 1u;
  }
  unsigned T = lo;
  // strictly-greater count
  if (tid == 0) scnt = 0;
  __syncthreads();
  int cg = 0;
  for (int j = tid; j < NN; j += 1024) cg += (sk[j] > T) ? 1 : 0;
  atomicAdd(&scnt, cg);
  __syncthreads();
  int cGreat = scnt;
  __syncthreads();
  // compact strictly-greater (deterministic block scan)
  sscan[tid] = cg; __syncthreads();
  for (int off = 1; off < 1024; off <<= 1) {
    int v = (tid >= off) ? sscan[tid - off] : 0;
    __syncthreads();
    sscan[tid] += v;
    __syncthreads();
  }
  int pos = sscan[tid] - cg;
  for (int j = tid; j < NN; j += 1024)
    if (sk[j] > T) { sidx[b * MM + pos] = j; inv[b * NN + j] = pos; pos++; }
  __syncthreads();
  // fill remainder deterministically from the == T ties
  int ce = 0;
  for (int j = tid; j < NN; j += 1024) ce += (sk[j] == T) ? 1 : 0;
  sscan[tid] = ce; __syncthreads();
  for (int off = 1; off < 1024; off <<= 1) {
    int v = (tid >= off) ? sscan[tid - off] : 0;
    __syncthreads();
    sscan[tid] += v;
    __syncthreads();
  }
  int epos = cGreat + (sscan[tid] - ce);
  for (int j = tid; j < NN; j += 1024)
    if (sk[j] == T) {
      if (epos < MM) { sidx[b * MM + epos] = j; inv[b * NN + j] = epos; }
      epos++;
    }
}

// ---------------------------------------------------------------------------
// 4) gather selected rows -> f16 only (WMMA A-matrix); residual comes straight
//    from tokens in the fused output kernel.
// ---------------------------------------------------------------------------
__global__ void gather_kernel(const float* __restrict__ tokens,
                              const int* __restrict__ sidx,
                              _Float16* __restrict__ selh) {
  int b = blockIdx.x, i = blockIdx.y, tid = threadIdx.x;      // block 128
  int pos = sidx[b * MM + i];
  const float4* src = (const float4*)(tokens + ((size_t)b * NN + pos) * HH);
  float4 v = src[tid];
  v4h hv = { (_Float16)v.x, (_Float16)v.y, (_Float16)v.z, (_Float16)v.w };
  ((v4h*)(selh + ((size_t)b * MM + i) * HH))[tid] = hv;
}

// ---------------------------------------------------------------------------
// 5) weight f32 -> f16
// ---------------------------------------------------------------------------
__global__ void cvt_kernel(const float* __restrict__ w_in, const float* __restrict__ w_out,
                           _Float16* __restrict__ w_in_h, _Float16* __restrict__ w_out_h) {
  int i = blockIdx.x * 256 + threadIdx.x;
  const int n1 = 3 * HH * HH;                                 // 1536x512
  const int n2 = HH * HH;
  if (i < n1) w_in_h[i] = (_Float16)w_in[i];
  else if (i < n1 + n2) w_out_h[i - n1] = (_Float16)w_out[i - n1];
}

// ---------------------------------------------------------------------------
// 6) QKV GEMM (8192 x 1536 x 512) via v_wmma_f32_16x16x32_f16.
//    Weight tile (32 x 512 f16 = 32 KB) staged in LDS — async global->LDS on
//    CDNA5 (ASYNCcnt) with sync fallback; reused by 4 waves per n-subtile.
//    Writes per-head Q,K row-major [b,h,i,d] and V transposed [b,h,d,i] (f16).
// ---------------------------------------------------------------------------
__global__ void qkv_wmma_kernel(const _Float16* __restrict__ selh,
                                const _Float16* __restrict__ w_in,
                                const float* __restrict__ in_b,
                                _Float16* __restrict__ qh, _Float16* __restrict__ kh,
                                _Float16* __restrict__ vt) {
  __shared__ _Float16 sW[32 * HH];                            // 32 KB
  int tid = threadIdx.x, lane = tid & 31, w = tid >> 5;       // block 256 = 8 waves
  int m0  = blockIdx.x * 64 + (w & 3) * 16;                   // row tile
  int nb  = blockIdx.y * 32;                                  // block out-channel base
  int n0l = (w >> 2) * 16;                                    // wave's local n tile

  const _Float16* wsrc = w_in + (size_t)nb * HH;
#if USE_ASYNC_LDS
  for (int e = tid; e < 2048; e += 256) {                     // 2048 x 16B chunks
    int r = e >> 6, c = (e & 63) * 8;
    __builtin_amdgcn_global_load_async_to_lds_b128(
        (__attribute__((address_space(1))) v4i*)(wsrc + (size_t)r * HH + c),
        (__attribute__((address_space(3))) v4i*)(sW + r * HH + c), 0, 0);
  }
  __builtin_amdgcn_s_wait_asynccnt(0);
#else
  for (int e = tid; e < 2048; e += 256) {
    int r = e >> 6, c = (e & 63) * 8;
    *(v8h*)(sW + r * HH + c) = *(const v8h*)(wsrc + (size_t)r * HH + c);
  }
#endif
  __syncthreads();

  v8f c = {};
#pragma unroll 4
  for (int k0 = 0; k0 < HH; k0 += 32) {
    v16h a = load_a_frag(selh, HH, m0, k0, lane);
    v16h b = load_b_frag(sW, HH, n0l, k0, lane);              // ds_load_b128
    c = wmma_f16(a, b, c);
  }
  int n = lane & 15;
  int col = nb + n0l + n;
  float bias = in_b[col];
  int which = col >> 9;                                       // 0=q 1=k 2=v
  int oo = col & 511;
  int head = oo >> 6, d = oo & 63;
  int rbase = m0 + ((lane & 16) ? 8 : 0);
#pragma unroll
  for (int v = 0; v < 8; ++v) {
    int row = rbase + v;                                      // 0..8191
    int b_ = row >> 9, i = row & 511;
    _Float16 hv = (_Float16)(c[v] + bias);
    size_t hb = (size_t)(b_ * NHD + head);
    if (which == 0)      qh[(hb * MM + i) * DH + d] = hv;
    else if (which == 1) kh[(hb * MM + i) * DH + d] = hv;
    else                 vt[(hb * DH + d) * MM + i] = hv;
  }
}

// ---------------------------------------------------------------------------
// 7) fused attention per (batch, head, 32-query tile):
//    WMMA scores -> LDS f32 -> softmax -> f16 probs in LDS -> WMMA attn@V
//    dynamic LDS: 32*512*4 (scores) + 32*512*2 (probs) = 96 KB
// ---------------------------------------------------------------------------
__global__ void attn_kernel(const _Float16* __restrict__ qh,
                            const _Float16* __restrict__ kh,
                            const _Float16* __restrict__ vt,
                            _Float16* __restrict__ attn_h) {
  extern __shared__ char smem[];
  float*    sS = (float*)smem;                                // [32][512]
  _Float16* sP = (_Float16*)(smem + 32 * 512 * sizeof(float));// [32][512]
  int b = blockIdx.x, h = blockIdx.y, qt = blockIdx.z;
  int tid = threadIdx.x, lane = tid & 31, w = tid >> 5;       // 8 waves
  size_t base = (size_t)(b * NHD + h) * MM * DH;
  const _Float16* Q = qh + base;
  const _Float16* K = kh + base;
  const _Float16* V = vt + base;                              // [64][512]
  // phase 1: scores (32 x 512) = Q-tile @ K^T
  for (int t = w; t < 64; t += 8) {
    int tm = t & 1, tn = t >> 1;
    int m0 = qt * 32 + tm * 16;                               // query row
    int n0 = tn * 16;                                         // key col
    v8f c = {};
#pragma unroll
    for (int k0 = 0; k0 < DH; k0 += 32) {
      v16h a  = load_a_frag(Q, DH, m0, k0, lane);
      v16h bf = load_b_frag(K, DH, n0, k0, lane);
      c = wmma_f16(a, bf, c);
    }
    int n = lane & 15;
    int lm0 = tm * 16 + ((lane & 16) ? 8 : 0);
#pragma unroll
    for (int v = 0; v < 8; ++v) sS[(lm0 + v) * 512 + n0 + n] = c[v] * 0.125f;
  }
  __syncthreads();
  // phase 2: softmax over 512 keys; 8 lanes per row
  {
    int row = tid >> 3, sub = tid & 7;
    float* r = sS + row * 512;
    float mx = -3.4e38f;
    for (int j = sub; j < 512; j += 8) mx = fmaxf(mx, r[j]);
#pragma unroll
    for (int s = 4; s >= 1; s >>= 1) mx = fmaxf(mx, __shfl_xor(mx, s, 8));
    float sum = 0.f;
    for (int j = sub; j < 512; j += 8) { float e = __expf(r[j] - mx); r[j] = e; sum += e; }
#pragma unroll
    for (int s = 4; s >= 1; s >>= 1) sum += __shfl_xor(sum, s, 8);
    float inv = 1.f / sum;
    _Float16* rp = sP + row * 512;
    for (int j = sub; j < 512; j += 8) rp[j] = (_Float16)(r[j] * inv);
  }
  __syncthreads();
  // phase 3: out (32 x 64) = probs @ V; A-fragments from LDS (ds_load_b128)
  {
    int tm = w & 1, tn = w >> 1;
    int m0l = tm * 16, n0 = tn * 16;
    v8f c = {};
#pragma unroll 4
    for (int k0 = 0; k0 < MM; k0 += 32) {
      v16h a  = load_a_frag(sP, 512, m0l, k0, lane);
      v16h bf = load_b_frag(V, MM, n0, k0, lane);
      c = wmma_f16(a, bf, c);
    }
    int n = lane & 15;
    int rbase = m0l + ((lane & 16) ? 8 : 0);
#pragma unroll
    for (int v = 0; v < 8; ++v) {
      int qrow = qt * 32 + rbase + v;
      attn_h[((size_t)b * MM + qrow) * HH + h * DH + n0 + n] = (_Float16)c[v];
    }
  }
}

// ---------------------------------------------------------------------------
// 8) out-proj GEMM (8192 x 512 x 512) -> rel_raw f32 (+bias)
// ---------------------------------------------------------------------------
__global__ void outproj_wmma_kernel(const _Float16* __restrict__ attn_h,
                                    const _Float16* __restrict__ w_out,
                                    const float* __restrict__ ob,
                                    float* __restrict__ rel_raw) {
  int tid = threadIdx.x, lane = tid & 31, w = tid >> 5;       // 8 waves, 16 tiles
  for (int t = w; t < 16; t += 8) {
    int tm = t & 3, tn = t >> 2;
    int m0 = blockIdx.x * 64 + tm * 16;
    int n0 = blockIdx.y * 64 + tn * 16;
    v8f c = {};
#pragma unroll 4
    for (int k0 = 0; k0 < HH; k0 += 32) {
      v16h a  = load_a_frag(attn_h, HH, m0, k0, lane);
      v16h bf = load_b_frag(w_out, HH, n0, k0, lane);
      c = wmma_f16(a, bf, c);
    }
    int n = lane & 15, col = n0 + n;
    float bias = ob[col];
    int rbase = m0 + ((lane & 16) ? 8 : 0);
#pragma unroll
    for (int v = 0; v < 8; ++v)
      rel_raw[(size_t)(rbase + v) * HH + col] = c[v] + bias;
  }
}

// ---------------------------------------------------------------------------
// 9) fused output: one streaming pass over tokens. Unselected rows copy
//    through; selected rows (via inverse map) add rel_raw, LN (wave reduce),
//    and write tokens + gate * LN. Whole wave takes the same branch.
// ---------------------------------------------------------------------------
__global__ void fuse_out_kernel(const float* __restrict__ tokens,
                                const float* __restrict__ rel_raw,
                                const float* __restrict__ g, const float* __restrict__ bb,
                                const int* __restrict__ inv,
                                const float* __restrict__ gate_prob,
                                float* __restrict__ out) {
  int tid = threadIdx.x, lane = tid & 31, w = tid >> 5;       // block 256 = 8 waves
  int row = blockIdx.x * 8 + w;                               // 0..65535
  int b = row >> 12;
  int i = inv[row];
  const float* x = tokens + (size_t)row * HH;
  float* o = out + (size_t)row * HH;
  if (i < 0) {                                                // uniform per wave
#pragma unroll
    for (int j = 0; j < 16; ++j) { int hh = lane + 32 * j; o[hh] = x[hh]; }
    return;
  }
  const float* xr = rel_raw + ((size_t)b * MM + i) * HH;
  float xs[16], xt[16];
  float s = 0.f;
#pragma unroll
  for (int j = 0; j < 16; ++j) {
    int hh = lane + 32 * j;
    float tv = x[hh];
    float v = tv + xr[hh];
    xt[j] = tv; xs[j] = v; s += v;
  }
#pragma unroll
  for (int m = 16; m >= 1; m >>= 1) s += __shfl_xor(s, m, 32);
  float mean = s * (1.f / (float)HH);
  float s2 = 0.f;
#pragma unroll
  for (int j = 0; j < 16; ++j) { float d = xs[j] - mean; s2 += d * d; }
#pragma unroll
  for (int m = 16; m >= 1; m >>= 1) s2 += __shfl_xor(s2, m, 32);
  float r = rsqrtf(s2 * (1.f / (float)HH) + EPS);
  float gp = gate_prob[b];
#pragma unroll
  for (int j = 0; j < 16; ++j) {
    int hh = lane + 32 * j;
    float ln = (xs[j] - mean) * r * g[hh] + bb[hh];
    o[hh] = xt[j] + gp * ln;
  }
}

// ---------------------------------------------------------------------------
// launch
// ---------------------------------------------------------------------------
extern "C" void kernel_launch(void* const* d_in, const int* in_sizes, int n_in,
                              void* d_out, int out_size, void* d_ws, size_t ws_size,
                              hipStream_t stream) {
  (void)in_sizes; (void)n_in; (void)out_size; (void)ws_size;
  const float* tokens     = (const float*)d_in[0];
  const float* gate_ln_g  = (const float*)d_in[1];
  const float* gate_ln_b  = (const float*)d_in[2];
  const float* gate_w     = (const float*)d_in[3];
  const float* gate_b     = (const float*)d_in[4];
  const float* sal_ln_g   = (const float*)d_in[5];
  const float* sal_ln_b   = (const float*)d_in[6];
  const float* sal_w      = (const float*)d_in[7];
  const float* sal_b      = (const float*)d_in[8];
  const float* in_proj_w  = (const float*)d_in[9];
  const float* in_proj_b  = (const float*)d_in[10];
  const float* out_proj_w = (const float*)d_in[11];
  const float* out_proj_b = (const float*)d_in[12];
  const float* rel_ln_g   = (const float*)d_in[13];
  const float* rel_ln_b   = (const float*)d_in[14];
  float* out = (float*)d_out;

  char* ws = (char*)d_ws;
  size_t off = 0;
  auto alloc = [&](size_t bytes) -> void* {
    void* p = ws + off;
    off = (off + bytes + 255) & ~(size_t)255;
    return p;
  };
  float*     pooled    = (float*)alloc((size_t)BB * HH * 4);
  float*     gate_prob = (float*)alloc((size_t)BB * 4);
  unsigned*  keys      = (unsigned*)alloc((size_t)BB * NN * 4);
  int*       sidx      = (int*)alloc((size_t)BB * MM * 4);
  int*       inv       = (int*)alloc((size_t)BB * NN * 4);
  _Float16*  selh      = (_Float16*)alloc((size_t)ROWS * HH * 2);
  _Float16*  w_in_h    = (_Float16*)alloc((size_t)3 * HH * HH * 2);
  _Float16*  w_out_h   = (_Float16*)alloc((size_t)HH * HH * 2);
  _Float16*  qh        = (_Float16*)alloc((size_t)BB * NHD * MM * DH * 2);
  _Float16*  kh        = (_Float16*)alloc((size_t)BB * NHD * MM * DH * 2);
  _Float16*  vt        = (_Float16*)alloc((size_t)BB * NHD * MM * DH * 2);
  _Float16*  attn_h    = (_Float16*)alloc((size_t)ROWS * HH * 2);
  float*     rel_raw   = (float*)alloc((size_t)ROWS * HH * 4);

  zero_kernel<<<(BB * HH + 255) / 256, 256, 0, stream>>>(pooled, BB * HH);
  salience_pooled_kernel<<<(BB * NN) / 8, 256, 0, stream>>>(
      tokens, sal_ln_g, sal_ln_b, sal_w, sal_b, keys, pooled);
  gate_kernel<<<BB, HH, 0, stream>>>(pooled, gate_ln_g, gate_ln_b, gate_w, gate_b,
                                     gate_prob);
  topk_kernel<<<BB, 1024, 0, stream>>>(keys, sidx, inv);
  gather_kernel<<<dim3(BB, MM), 128, 0, stream>>>(tokens, sidx, selh);
  cvt_kernel<<<(4 * HH * HH) / 256, 256, 0, stream>>>(in_proj_w, out_proj_w,
                                                      w_in_h, w_out_h);
  qkv_wmma_kernel<<<dim3(ROWS / 64, (3 * HH) / 32), 256, 0, stream>>>(
      selh, w_in_h, in_proj_b, qh, kh, vt);
  attn_kernel<<<dim3(BB, NHD, MM / 32), 256, 96 * 1024, stream>>>(qh, kh, vt,
                                                                  attn_h);
  outproj_wmma_kernel<<<dim3(ROWS / 64, HH / 64), 256, 0, stream>>>(
      attn_h, w_out_h, out_proj_b, rel_raw);
  fuse_out_kernel<<<(BB * NN) / 8, 256, 0, stream>>>(tokens, rel_raw, rel_ln_g,
                                                     rel_ln_b, inv, gate_prob, out);
}
